// AudioEncoderWrapper_82051055223096
// MI455X (gfx1250) — matrix operand order repaired and checked by reading, not compiled, not run on hardware
//
#include <hip/hip_runtime.h>
#include <hip/hip_bf16.h>

typedef __attribute__((ext_vector_type(16))) _Float16 v16h;
typedef __attribute__((ext_vector_type(8)))  _Float16 v8h;
typedef __attribute__((ext_vector_type(8)))  float    v8f;

#define NQ    16
#define CBS   2048
#define KDIM  64
#define DDIM  512
#define CINC  512
#define BATCH 8
#define TLEN  2048
#define NTOK  (BATCH * TLEN)   // 16384

#define WMMA_F16(a, b, c) \
  __builtin_amdgcn_wmma_f32_16x16x32_f16(false, (a), false, (b), (short)0, (c), false, false)

// Async copy of 16 bytes global -> LDS (per lane).
// Pass the LDS *pointer*: (a) ptrtoint captures the allocation so the
// "memory" clobber forces the compiler to assume the asm writes it (keeps the
// LDS reads live); (b) the low 32 bits of a generic pointer into LDS are the
// wave-relative LDS byte offset that VDST expects ({SHARED_BASE, as3_offset}).
#define ASYNC_G2L_B128(ldsptr, gptr)                                       \
  asm volatile("global_load_async_to_lds_b128 %0, %1, off"                 \
               :: "v"((unsigned)(unsigned long long)(const void*)(ldsptr)),\
                  "v"(gptr)                                                \
               : "memory")
#define WAIT_ASYNC() asm volatile("s_wait_asynccnt 0x0" ::: "memory")

// ---------------------------------------------------------------------------
// One-time conversions: weights f32->f16, codebook row-normalize -> f16.
// ---------------------------------------------------------------------------
__global__ void k_f32_to_f16(const float* __restrict__ src,
                             _Float16* __restrict__ dst, int n) {
  int i = blockIdx.x * blockDim.x + threadIdx.x;
  if (i < n) dst[i] = (_Float16)src[i];
}

__global__ void k_cbnorm(const float* __restrict__ cb,
                         _Float16* __restrict__ cbn) {
  int row = blockIdx.x * blockDim.x + threadIdx.x;   // NQ*CBS rows of 64
  if (row >= NQ * CBS) return;
  const float* s = cb + (size_t)row * KDIM;
  float acc = 0.f;
  for (int i = 0; i < KDIM; ++i) acc += s[i] * s[i];
  float inv = 1.0f / fmaxf(sqrtf(acc), 1e-12f);
  _Float16* d = cbn + (size_t)row * KDIM;
  for (int i = 0; i < KDIM; ++i) d[i] = (_Float16)(s[i] * inv);
}

// ---------------------------------------------------------------------------
// input_proj: R[tok][d] = sum_c W[d][c] * h[b][c][t] + bias[d]
// 4 waves / block (4 d-tiles); h tile (32c x 16t) double-buffered via async
// global->LDS; two independent WMMA accumulation chains (even/odd k-blocks).
// ---------------------------------------------------------------------------
__global__ void k_input_proj(const float* __restrict__ h,
                             const _Float16* __restrict__ w16,  // [512][512]
                             const float* __restrict__ bias,
                             float* __restrict__ R) {
  __shared__ float lb[2][32 * 16];
  const int tok0 = blockIdx.x * 16;
  const int b    = tok0 >> 11;
  const int t0   = tok0 & (TLEN - 1);
  const int wave = threadIdx.x >> 5;
  const int lane = threadIdx.x & 31;
  const int nl   = lane & 15;
  const int hi   = lane >> 4;
  const int m0   = blockIdx.y * 64 + wave * 16;

  const int cl = threadIdx.x >> 2;                 // 0..31 (c row)
  const int tq = (threadIdx.x & 3) * 4;            // 0,4,8,12 (t chunk)
  const int lpos = cl * 16 + tq;                   // element offset in a tile

  {  // prologue: stage tile 0 into buffer 0
    const float* gp = h + ((size_t)b * CINC + cl) * TLEN + t0 + tq;
    ASYNC_G2L_B128(&lb[0][lpos], gp);
  }

  v8f acc0 = {}, acc1 = {};
#pragma unroll 2
  for (int it = 0; it < 16; ++it) {
    const int k0 = it * 32;
    WAIT_ASYNC();
    __syncthreads();   // tile `it` fully staged; everyone done with other buf
    if (it + 1 < 16) {
      const float* gp = h + ((size_t)b * CINC + k0 + 32 + cl) * TLEN + t0 + tq;
      ASYNC_G2L_B128(&lb[(it + 1) & 1][lpos], gp);
    }
    const float* cur = &lb[it & 1][0];
    const _Float16* ap = w16 + (size_t)(m0 + nl) * CINC + k0 + hi * 16;
    v16h a, bm;
#pragma unroll
    for (int i = 0; i < 8; ++i) { a[i] = ap[i]; a[8 + i] = ap[8 + i]; }
#pragma unroll
    for (int j = 0; j < 16; ++j) bm[j] = (_Float16)cur[(hi * 16 + j) * 16 + nl];
    if (it & 1) acc1 = WMMA_F16(a, bm, acc1);
    else        acc0 = WMMA_F16(a, bm, acc0);
  }
  const v8f acc = acc0 + acc1;
  const int dbase = m0 + hi * 8;
  float* out = R + (size_t)(tok0 + nl) * DDIM + dbase;
#pragma unroll
  for (int r = 0; r < 8; ++r) out[r] = acc[r] + bias[dbase + r];
}

// ---------------------------------------------------------------------------
// in_proj: ZE[tok][k] = f16( sum_d Win[k][d] * mask(t)*R[tok][d] + b_in[k] )
// Residual tile (16tok x 32d) double-buffered async; LDS stride 36 (16 banks).
// Mask applied at B-fragment build (per-lane uniform select).
// ---------------------------------------------------------------------------
__global__ void k_in_proj(const float* __restrict__ R,
                          const _Float16* __restrict__ win16, // [NQ][64][512]
                          const float* __restrict__ b_in,     // [NQ][64]
                          const int* __restrict__ lens,
                          _Float16* __restrict__ ZE,          // [NTOK][64]
                          int q) {
  __shared__ float lb[2][16 * 36];
  const int tok0 = blockIdx.x * 16;
  const int b    = tok0 >> 11;
  const int t0   = tok0 & (TLEN - 1);
  const int len  = lens[b];
  const int wave = threadIdx.x >> 5;
  const int lane = threadIdx.x & 31;
  const int nl   = lane & 15;
  const int hi   = lane >> 4;
  const int m0   = wave * 16;
  const _Float16* W = win16 + (size_t)q * KDIM * DDIM;

  const int tl = threadIdx.x >> 3;                 // 0..15 (token row)
  const int cq = (threadIdx.x & 7) * 4;            // 0..28 (d chunk)
  const int lpos = tl * 36 + cq;
  const bool tv = (t0 + nl) < len;

  {  // prologue
    const float* gp = R + (size_t)(tok0 + tl) * DDIM + cq;
    ASYNC_G2L_B128(&lb[0][lpos], gp);
  }

  v8f acc0 = {}, acc1 = {};
#pragma unroll 2
  for (int it = 0; it < 16; ++it) {
    const int k0 = it * 32;
    WAIT_ASYNC();
    __syncthreads();
    if (it + 1 < 16) {
      const float* gp = R + (size_t)(tok0 + tl) * DDIM + k0 + 32 + cq;
      ASYNC_G2L_B128(&lb[(it + 1) & 1][lpos], gp);
    }
    const float* cur = &lb[it & 1][0];
    const _Float16* ap = W + (size_t)(m0 + nl) * DDIM + k0 + hi * 16;
    v16h a, bm;
#pragma unroll
    for (int i = 0; i < 8; ++i) { a[i] = ap[i]; a[8 + i] = ap[8 + i]; }
#pragma unroll
    for (int j = 0; j < 16; ++j) {
      float v = cur[nl * 36 + hi * 16 + j];
      bm[j] = tv ? (_Float16)v : (_Float16)0.f;
    }
    if (it & 1) acc1 = WMMA_F16(a, bm, acc1);
    else        acc0 = WMMA_F16(a, bm, acc0);
  }
  const v8f acc = acc0 + acc1;
  const int kb = m0 + hi * 8;
  const float* bi = b_in + q * KDIM + kb;
  v8h o;
#pragma unroll
  for (int r = 0; r < 8; ++r) o[r] = (_Float16)(acc[r] + bi[r]);
  *(v8h*)(ZE + (size_t)(tok0 + nl) * KDIM + kb) = o;
}

// ---------------------------------------------------------------------------
// scores + argmax: idx[tok] = argmax_c ZE[tok][:] . CBN[c][:]
// One wave owns 2 token tiles (32 tokens): A resident in regs, each B tile
// feeds 4 WMMAs over 2 independent accumulation chains. Ties -> lowest index.
// ---------------------------------------------------------------------------
__global__ void k_scores(const _Float16* __restrict__ ZE,
                         const _Float16* __restrict__ CBN, // [NQ][CBS][64]
                         int* __restrict__ idx_i,
                         float* __restrict__ idx_f,        // d_out slice
                         int q) {
  const int wave = threadIdx.x >> 5;
  const int lane = threadIdx.x & 31;
  const int nl   = lane & 15;
  const int hi   = lane >> 4;
  const int tok0 = (blockIdx.x * 4 + wave) * 32;

  v16h A0[2], A1[2];
#pragma unroll
  for (int s = 0; s < 2; ++s) {
    const _Float16* ap = ZE + (size_t)(tok0 + s * 16 + nl) * KDIM;
#pragma unroll
    for (int i = 0; i < 8; ++i) {
      A0[s][i]     = ap[hi * 16 + i];       A0[s][8 + i] = ap[hi * 16 + 8 + i];
      A1[s][i]     = ap[32 + hi * 16 + i];  A1[s][8 + i] = ap[32 + hi * 16 + 8 + i];
    }
  }
  const _Float16* C = CBN + (size_t)q * CBS * KDIM;

  float bestv[2][8]; int besti[2][8];
#pragma unroll
  for (int s = 0; s < 2; ++s)
#pragma unroll
    for (int r = 0; r < 8; ++r) { bestv[s][r] = -3.0e38f; besti[s][r] = 0; }

  for (int c0 = 0; c0 < CBS; c0 += 16) {
    const _Float16* crow = C + (size_t)(c0 + nl) * KDIM;  // this lane's column
    __builtin_prefetch((const char*)crow + 32 * KDIM * 2, 0, 0);  // +2 tiles
    v16h b0 = *(const v16h*)(crow + hi * 16);
    v16h b1 = *(const v16h*)(crow + 32 + hi * 16);
    const int code = c0 + nl;
#pragma unroll
    for (int s = 0; s < 2; ++s) {
      v8f acc = {};
      acc = WMMA_F16(A0[s], b0, acc);
      acc = WMMA_F16(A1[s], b1, acc);
#pragma unroll
      for (int r = 0; r < 8; ++r)
        if (acc[r] > bestv[s][r]) { bestv[s][r] = acc[r]; besti[s][r] = code; }
    }
  }
#pragma unroll
  for (int s = 0; s < 2; ++s) {
#pragma unroll
    for (int r = 0; r < 8; ++r) {
      float v = bestv[s][r]; int ii = besti[s][r];
      for (int m = 8; m >= 1; m >>= 1) {
        float ov = __shfl_xor(v, m, 16);
        int   oi = __shfl_xor(ii, m, 16);
        if (ov > v || (ov == v && oi < ii)) { v = ov; ii = oi; }
      }
      bestv[s][r] = v; besti[s][r] = ii;
    }
    if (nl == 0) {
      const int tokb = tok0 + s * 16 + hi * 8;
#pragma unroll
      for (int r = 0; r < 8; ++r) {
        idx_i[tokb + r] = besti[s][r];
        idx_f[tokb + r] = (float)besti[s][r];
      }
    }
  }
}

// ---------------------------------------------------------------------------
// gather + out_proj + residual update:
// R[tok][d] -= mask(t) * ( sum_k Wout[d][k]*cb[idx[tok]][k] + b_out[d] )
// Indexed codebook gather via async global->LDS b128 (f32 rows, stride 68 ->
// bank offset 4*nl+c, conflict-free reads), converted to f16 at B build.
// ---------------------------------------------------------------------------
__global__ void k_update(const float* __restrict__ cb,       // [NQ][CBS][64]
                         const _Float16* __restrict__ wout16,// [NQ][512][64]
                         const float* __restrict__ b_out,    // [NQ][512]
                         const int* __restrict__ idx_i,
                         const int* __restrict__ lens,
                         float* __restrict__ R,
                         int q) {
  __shared__ float zq[16 * 68];
  const int tok0 = blockIdx.x * 16;
  const int b    = tok0 >> 11;
  const int t0   = tok0 & (TLEN - 1);
  const int len  = lens[b];
  const int wave = threadIdx.x >> 5;
  const int lane = threadIdx.x & 31;
  const int nl   = lane & 15;
  const int hi   = lane >> 4;
  const float* CB = cb + (size_t)q * CBS * KDIM;

  {
    const int tl = threadIdx.x >> 4;               // 0..15 (token)
    const int cq = (threadIdx.x & 15) * 4;         // 0..60 (k chunk)
    const float* gp = CB + (size_t)idx_i[tok0 + tl] * KDIM + cq;
    ASYNC_G2L_B128(&zq[tl * 68 + cq], gp);         // async indexed gather
    WAIT_ASYNC();
  }
  __syncthreads();

  const float* zr = zq + nl * 68;
  v16h b0, b1;
#pragma unroll
  for (int j = 0; j < 16; ++j) {
    b0[j] = (_Float16)zr[hi * 16 + j];
    b1[j] = (_Float16)zr[32 + hi * 16 + j];
  }
  const bool valid = (t0 + nl) < len;
  const _Float16* W = wout16 + (size_t)q * DDIM * KDIM;

  for (int j = 0; j < 4; ++j) {
    const int m0 = (wave * 4 + j) * 16;
    const _Float16* ap = W + (size_t)(m0 + nl) * KDIM;
    v16h a0, a1;
#pragma unroll
    for (int i = 0; i < 8; ++i) {
      a0[i]     = ap[hi * 16 + i];       a0[8 + i] = ap[hi * 16 + 8 + i];
      a1[i]     = ap[32 + hi * 16 + i];  a1[8 + i] = ap[32 + hi * 16 + 8 + i];
    }
    v8f acc = {};
    acc = WMMA_F16(a0, b0, acc);
    acc = WMMA_F16(a1, b1, acc);
    const int dbase = m0 + hi * 8;
    float* rp = R + (size_t)(tok0 + nl) * DDIM + dbase;
    const float* bo = b_out + q * DDIM + dbase;
    if (valid)
#pragma unroll
      for (int r = 0; r < 8; ++r) rp[r] -= (acc[r] + bo[r]);
  }
}

// ---------------------------------------------------------------------------
extern "C" void kernel_launch(void* const* d_in, const int* in_sizes, int n_in,
                              void* d_out, int out_size, void* d_ws,
                              size_t ws_size, hipStream_t stream) {
  (void)in_sizes; (void)n_in; (void)out_size; (void)ws_size;
  const float* h     = (const float*)d_in[0];
  const int*   lens  = (const int*)d_in[1];
  const float* w_ip  = (const float*)d_in[2];
  const float* b_ip  = (const float*)d_in[3];
  const float* w_in  = (const float*)d_in[4];
  const float* b_in  = (const float*)d_in[5];
  const float* cb    = (const float*)d_in[6];
  const float* w_out = (const float*)d_in[7];
  const float* b_out = (const float*)d_in[8];
  float* outf = (float*)d_out;

  char* ws = (char*)d_ws;
  size_t off = 0;
  float*    R    = (float*)(ws + off);    off += (size_t)NTOK * DDIM * 4;
  _Float16* ZE   = (_Float16*)(ws + off); off += (size_t)NTOK * KDIM * 2;
  int*      IDX  = (int*)(ws + off);      off += (size_t)NTOK * 4;
  _Float16* WIP  = (_Float16*)(ws + off); off += (size_t)DDIM * CINC * 2;
  _Float16* WIN  = (_Float16*)(ws + off); off += (size_t)NQ * KDIM * DDIM * 2;
  _Float16* WOUT = (_Float16*)(ws + off); off += (size_t)NQ * DDIM * KDIM * 2;
  _Float16* CBN  = (_Float16*)(ws + off); off += (size_t)NQ * CBS * KDIM * 2;

  k_f32_to_f16<<<(DDIM * CINC + 255) / 256, 256, 0, stream>>>(w_ip, WIP, DDIM * CINC);
  k_f32_to_f16<<<(NQ * KDIM * DDIM + 255) / 256, 256, 0, stream>>>(w_in, WIN, NQ * KDIM * DDIM);
  k_f32_to_f16<<<(NQ * DDIM * KDIM + 255) / 256, 256, 0, stream>>>(w_out, WOUT, NQ * DDIM * KDIM);
  k_cbnorm<<<(NQ * CBS + 255) / 256, 256, 0, stream>>>(cb, CBN);

  k_input_proj<<<dim3(NTOK / 16, DDIM / 64), 128, 0, stream>>>(h, WIP, b_ip, R);

  for (int q = 0; q < NQ; ++q) {
    k_in_proj<<<NTOK / 16, 128, 0, stream>>>(R, WIN, b_in, lens, ZE, q);
    k_scores<<<NTOK / 128, 128, 0, stream>>>(ZE, CBN, IDX,
                                             outf + (size_t)q * NTOK, q);
    k_update<<<NTOK / 16, 256, 0, stream>>>(cb, WOUT, b_out, IDX, lens, R, q);
  }
}
